// PerformerEncoderLayer_16449724744408
// MI455X (gfx1250) — compile-verified, matrix operand order checked by reading, not run on hardware
//
#include <hip/hip_runtime.h>
#include <hip/hip_bf16.h>
#include <math.h>

// ---------------- CDNA5 WMMA plumbing (wave32, 16x16x32 bf16) ----------------
typedef __attribute__((ext_vector_type(16))) __bf16 bf16x16;
typedef __attribute__((ext_vector_type(8)))  float  f32x8;

__device__ __forceinline__ unsigned short f2bf(float f) {
  unsigned int u = __float_as_uint(f);
  u += 0x7FFFu + ((u >> 16) & 1u);            // RNE
  return (unsigned short)(u >> 16);
}
__device__ __forceinline__ float bf2f(unsigned short h) {
  return __uint_as_float(((unsigned int)h) << 16);
}

union FragBF { bf16x16 v; uint4 q[2]; };

// Fragment loader for A (MxK row-major) and for B supplied as Wt (NxK row-major).
// ISA layout: lane r=lane&15 -> row; halves [0..7] at k0+8g, [8..15] at k0+16+8g.
// ld must be a multiple of 8 (16B alignment of each b128 load).
__device__ __forceinline__ bf16x16 load_frag(const unsigned short* base, int ld, int lane) {
  const int g = lane >> 4, r = lane & 15;
  const unsigned short* p = base + (size_t)r * ld + g * 8;
  FragBF f;
  f.q[0] = *(const uint4*)(p);
  f.q[1] = *(const uint4*)(p + 16);
  return f.v;
}

__device__ __forceinline__ f32x8 wmma_bf16(bf16x16 a, bf16x16 b, f32x8 c) {
  return __builtin_amdgcn_wmma_f32_16x16x32_bf16(false, a, false, b, (short)0, c, false, false);
}
__device__ __forceinline__ f32x8 zero8() {
  f32x8 z = {0.f,0.f,0.f,0.f,0.f,0.f,0.f,0.f}; return z;
}

// ---------------- CDNA5 async global->LDS copy (ASYNCcnt path) ----------------
// GLOBAL_LOAD_ASYNC_TO_LDS_B128, GV mode: vdst = per-lane LDS byte offset,
// vaddr = 64-bit global address. Generic LDS address low 32 bits == LDS offset
// (aperture occupies the high half), so truncation yields the DS address.
__device__ __forceinline__ void async_copy_b128(const void* gptr, void* lptr) {
  unsigned lds = (unsigned)(uintptr_t)lptr;
  unsigned long long ga = (unsigned long long)(uintptr_t)gptr;
  asm volatile("global_load_async_to_lds_b128 %0, %1, off"
               :: "v"(lds), "v"(ga)
               : "memory");
}
__device__ __forceinline__ void wait_async0() {
#if __has_builtin(__builtin_amdgcn_s_wait_asynccnt)
  __builtin_amdgcn_s_wait_asynccnt(0);
#else
  asm volatile("s_wait_asynccnt 0x0" ::: "memory");
#endif
}

// sign-split float atomic max (works on int-typed storage holding float bits)
__device__ __forceinline__ void atomicMaxFloatBits(int* addr, float v) {
  if (v >= 0.f) atomicMax(addr, __float_as_int(v));
  else          atomicMin((unsigned int*)addr, (unsigned int)__float_as_int(v));
}

// ---------------- problem constants ----------------
constexpr int kB = 4, kS = 4096, kDM = 1024, kNH = 16, kDH = 64, kDFF = 4096;
constexpr int kMF = 266;        // real random features
constexpr int kMP = 320;        // padded features (mult of 32)
constexpr int kLDT = 72;        // LDS ld for 64-wide tiles (mult of 8, dodges conflicts)
constexpr int kLDQ = 328;       // LDS ld for qp rows (mult of 8)
constexpr long kROWS = (long)kB * kS;       // 16384

// ---------------- elementwise converts ----------------
__global__ void f32bf_kernel(const float* __restrict__ in, unsigned short* __restrict__ out, int n) {
  int i = blockIdx.x * 256 + threadIdx.x;
  if (i < n) out[i] = f2bf(in[i]);
}

__global__ void proj_kernel(const float* __restrict__ proj, unsigned short* __restrict__ projBF) {
  int i = blockIdx.x * 256 + threadIdx.x;   // kMP*64
  if (i >= kMP * 64) return;
  int m = i >> 6, d = i & 63;
  const float dn = 0.3535533906f;           // 64^-0.25
  projBF[i] = (m < kMF) ? f2bf(dn * proj[m * 64 + d]) : (unsigned short)0;
}

__global__ void init_kernel(float* __restrict__ ctx, float* __restrict__ ksum,
                            float* __restrict__ kmax, int nctx, int nsum) {
  int i = blockIdx.x * 256 + threadIdx.x;
  if (i < nctx) ctx[i] = 0.f;
  if (i < nsum) ksum[i] = 0.f;
  if (i < 64)   kmax[i] = __uint_as_float(0xFF800000u); // -inf
}

// ---------------- generic bf16 WMMA GEMM: C = A(MxK) * Wt(NxK)^T + bias ----------------
// Double-buffered LDS with async global->LDS prefetch pipeline.
// flags: 1=relu, 2=write f32, 4=write bf16
__global__ __launch_bounds__(256) void gemm_bf16(
    const unsigned short* __restrict__ A, const unsigned short* __restrict__ Bw,
    const float* __restrict__ bias, float* __restrict__ Cf, unsigned short* __restrict__ Cb,
    int M, int N, int K, int lda, int ldb, int ldc, int flags)
{
  constexpr int BM = 128, BN = 64, BK = 32;
  __shared__ unsigned short As[2][BM * BK];
  __shared__ unsigned short Bs[2][BN * BK];
  const int tid = threadIdx.x, lane = tid & 31, wid = tid >> 5;
  const int wm = wid & 3, wn = wid >> 2;     // 4x2 waves, each 32x32
  const int m0 = blockIdx.y * BM, n0 = blockIdx.x * BN;
  const int g = lane >> 4, lc = lane & 15;

  f32x8 acc[2][2];
  for (int i = 0; i < 2; ++i) for (int j = 0; j < 2; ++j) acc[i][j] = zero8();

  auto stage = [&](int k0, int bufIdx) {
    for (int i = tid; i < BM * 4; i += 256) {
      int row = i >> 2, seg = i & 3;
      async_copy_b128(A + (size_t)(m0 + row) * lda + k0 + seg * 8,
                      &As[bufIdx][row * BK + seg * 8]);
    }
    for (int i = tid; i < BN * 4; i += 256) {
      int row = i >> 2, seg = i & 3;
      async_copy_b128(Bw + (size_t)(n0 + row) * ldb + k0 + seg * 8,
                      &Bs[bufIdx][row * BK + seg * 8]);
    }
  };

  stage(0, 0);
  wait_async0();
  __syncthreads();

  int buf = 0;
  for (int k0 = 0; k0 < K; k0 += BK) {
    if (k0 + BK < K) stage(k0 + BK, buf ^ 1);   // prefetch next tile (async engine)
    #pragma unroll
    for (int mi = 0; mi < 2; ++mi) {
      bf16x16 a = load_frag(&As[buf][(wm * 32 + mi * 16) * BK], BK, lane);
      #pragma unroll
      for (int ni = 0; ni < 2; ++ni) {
        bf16x16 b = load_frag(&Bs[buf][(wn * 32 + ni * 16) * BK], BK, lane);
        acc[mi][ni] = wmma_bf16(a, b, acc[mi][ni]);
      }
    }
    wait_async0();      // own prefetch landed
    __syncthreads();    // all waves' prefetch landed, all reads of buf done
    buf ^= 1;
  }
  #pragma unroll
  for (int mi = 0; mi < 2; ++mi)
    #pragma unroll
    for (int ni = 0; ni < 2; ++ni) {
      int col = n0 + wn * 32 + ni * 16 + lc;
      float bv = bias ? bias[col] : 0.f;
      #pragma unroll
      for (int r = 0; r < 8; ++r) {
        int row = m0 + wm * 32 + mi * 16 + r + 8 * g;
        float v = acc[mi][ni][r] + bv;
        if (flags & 1) v = fmaxf(v, 0.f);
        if (flags & 2) Cf[(size_t)row * ldc + col] = v;
        if (flags & 4) Cb[(size_t)row * ldc + col] = f2bf(v);
      }
    }
}

// ---------------- diag = 0.5*dn^2*|x|^2 per (b,h,s), layout [bh][s] ----------------
__global__ __launch_bounds__(256) void diag_kernel(const unsigned short* __restrict__ xBF,
                                                   float* __restrict__ diag) {
  int i = blockIdx.x * 256 + threadIdx.x;  // [0, 64*4096)
  int bh = i >> 12, s = i & 4095;
  int b = bh >> 4, h = bh & 15;
  const unsigned short* p = xBF + ((size_t)(b * kS + s)) * kDM + h * kDH;
  float acc = 0.f;
  #pragma unroll 8
  for (int d = 0; d < kDH; ++d) { float v = bf2f(p[d]); acc += v * v; }
  diag[i] = 0.0625f * acc;                 // 0.5 * 64^-0.5
}

// ---------------- global max over (s,m) of xd_k per (b,h) ----------------
__global__ __launch_bounds__(256) void kmax_kernel(const unsigned short* __restrict__ kBF,
                                                   const unsigned short* __restrict__ projBF,
                                                   float* __restrict__ kmax) {
  const int bh = blockIdx.x, b = bh >> 4, h = bh & 15;
  const int s = blockIdx.y * 256 + threadIdx.x;
  const unsigned short* p = kBF + ((size_t)(b * kS + s)) * kDM + h * kDH;
  float kv[kDH];
  #pragma unroll 8
  for (int d = 0; d < kDH; ++d) kv[d] = bf2f(p[d]);
  float mx = -3.0e38f;
  for (int m = 0; m < kMF; ++m) {
    const unsigned short* pm = projBF + m * kDH;
    float dot = 0.f;
    #pragma unroll 8
    for (int d = 0; d < kDH; ++d) dot += kv[d] * bf2f(pm[d]);
    mx = fmaxf(mx, dot);
  }
  __shared__ float red[256];
  red[threadIdx.x] = mx; __syncthreads();
  for (int st = 128; st > 0; st >>= 1) {
    if (threadIdx.x < st) red[threadIdx.x] = fmaxf(red[threadIdx.x], red[threadIdx.x + st]);
    __syncthreads();
  }
  if (threadIdx.x == 0) atomicMaxFloatBits((int*)&kmax[bh], red[0]);
}

// ---------------- fused: kp = phi(k); ctx += kp^T v; k_sum += col-sums ----------------
__global__ __launch_bounds__(320) void ctx_kernel(
    const unsigned short* __restrict__ kBF, const unsigned short* __restrict__ vBF,
    const unsigned short* __restrict__ projBF, const float* __restrict__ diagK,
    const float* __restrict__ kmax, float* __restrict__ ctx, float* __restrict__ ksum)
{
  __shared__ unsigned short kpT[kMP * kLDT];   // feature-major kp tile (320 x 64)
  __shared__ unsigned short tile[64 * kLDT];   // k tile, then v^T tile
  const int tid = threadIdx.x, lane = tid & 31, w = tid >> 5;
  const int bh = blockIdx.x, b = bh >> 4, h = bh & 15;
  const int g = lane >> 4, lc = lane & 15;
  const float km = kmax[bh];
  const float ratio = rsqrtf((float)kMF);
  f32x8 cacc[2][4];
  for (int i = 0; i < 2; ++i) for (int j = 0; j < 4; ++j) cacc[i][j] = zero8();
  float ksumLoc = 0.f;

  for (int it = 0; it < 8; ++it) {
    const int s0 = blockIdx.y * 512 + it * 64;
    const size_t grow = (size_t)b * kS + s0;
    // k tile -> LDS via async-to-LDS b128 (16B segments, all aligned)
    for (int i = tid; i < 64 * 8; i += 320) {
      int r = i >> 3, seg = i & 7;
      async_copy_b128(kBF + (grow + r) * kDM + h * kDH + seg * 8,
                      &tile[r * kLDT + seg * 8]);
    }
    wait_async0();
    __syncthreads();
    // xd = ktile(64x64) @ proj^T for feature cols [w*32, w*32+32)
    f32x8 x[4][2];
    for (int mt = 0; mt < 4; ++mt) for (int nt = 0; nt < 2; ++nt) x[mt][nt] = zero8();
    #pragma unroll
    for (int k0 = 0; k0 < 64; k0 += 32) {
      #pragma unroll
      for (int mt = 0; mt < 4; ++mt) {
        bf16x16 a = load_frag(tile + (mt * 16) * kLDT + k0, kLDT, lane);
        #pragma unroll
        for (int nt = 0; nt < 2; ++nt) {
          bf16x16 bq = load_frag(projBF + (size_t)(w * 32 + nt * 16) * kDH + k0, kDH, lane);
          x[mt][nt] = wmma_bf16(a, bq, x[mt][nt]);
        }
      }
    }
    // kp epilogue -> kpT (transposed store)
    #pragma unroll
    for (int mt = 0; mt < 4; ++mt)
      #pragma unroll
      for (int nt = 0; nt < 2; ++nt) {
        int feat = w * 32 + nt * 16 + lc;
        #pragma unroll
        for (int r = 0; r < 8; ++r) {
          int srow = mt * 16 + r + 8 * g;
          float kp = 0.f;
          if (feat < kMF)
            kp = ratio * (__expf(x[mt][nt][r] - diagK[bh * kS + s0 + srow] - km) + 1e-4f);
          kpT[feat * kLDT + srow] = f2bf(kp);
        }
      }
    __syncthreads();
    // v^T -> LDS (overwrites k tile); k_sum partials
    for (int i = tid; i < 64 * 64; i += 320) {
      int r = i >> 6, d = i & 63;
      tile[d * kLDT + r] = vBF[(grow + r) * kDM + h * kDH + d];
    }
    {
      float sacc = 0.f;
      #pragma unroll 8
      for (int ss = 0; ss < 64; ++ss) sacc += bf2f(kpT[tid * kLDT + ss]);
      ksumLoc += sacc;
    }
    __syncthreads();
    // ctx += kpT(320x64) @ vT(64x64)^T  (K-dim = s)
    #pragma unroll
    for (int k0 = 0; k0 < 64; k0 += 32) {
      #pragma unroll
      for (int mt = 0; mt < 2; ++mt) {
        bf16x16 a = load_frag(kpT + (size_t)(w * 32 + mt * 16) * kLDT + k0, kLDT, lane);
        #pragma unroll
        for (int nt = 0; nt < 4; ++nt) {
          bf16x16 bv = load_frag(tile + (nt * 16) * kLDT + k0, kLDT, lane);
          cacc[mt][nt] = wmma_bf16(a, bv, cacc[mt][nt]);
        }
      }
    }
    __syncthreads();
  }
  // accumulate to global
  #pragma unroll
  for (int mt = 0; mt < 2; ++mt)
    #pragma unroll
    for (int nt = 0; nt < 4; ++nt) {
      int dcol = nt * 16 + lc;
      #pragma unroll
      for (int r = 0; r < 8; ++r) {
        int feat = w * 32 + mt * 16 + r + 8 * g;
        atomicAdd(&ctx[((size_t)bh * kMP + feat) * kDH + dcol], cacc[mt][nt][r]);
      }
    }
  atomicAdd(&ksum[bh * kMP + tid], ksumLoc);
}

// ---------------- ctx fp32 -> ctxT bf16 (d-major for B fragments) ----------------
__global__ void ctxT_kernel(const float* __restrict__ ctx, unsigned short* __restrict__ ctxT) {
  int i = blockIdx.x * 256 + threadIdx.x;    // 64*320*64
  if (i >= 64 * kMP * kDH) return;
  int bh = i / (kMP * kDH); int rem = i - bh * (kMP * kDH);
  int m = rem >> 6, d = rem & 63;
  ctxT[((size_t)bh * kDH + d) * kMP + m] = f2bf(ctx[i]);
}

// ---------------- fused: qp = phi(q); out = (qp @ ctx) / (qp . k_sum) ----------------
__global__ __launch_bounds__(320) void attn_kernel(
    const unsigned short* __restrict__ qBF, const unsigned short* __restrict__ projBF,
    const unsigned short* __restrict__ ctxT, const float* __restrict__ diagQ,
    const float* __restrict__ ksum, unsigned short* __restrict__ attnBF)
{
  __shared__ unsigned short qt[64 * kLDT];
  __shared__ unsigned short qps[64 * kLDQ];
  __shared__ int   rowmax[64];
  __shared__ float ksumS[kMP];
  __shared__ float dinv[64];
  const int tid = threadIdx.x, lane = tid & 31, w = tid >> 5;
  const int bh = blockIdx.x, b = bh >> 4, h = bh & 15;
  const int g = lane >> 4, lc = lane & 15;
  const int s0 = blockIdx.y * 64;
  const size_t grow = (size_t)b * kS + s0;
  const float ratio = rsqrtf((float)kMF);

  // q tile -> LDS via async-to-LDS b128
  for (int i = tid; i < 64 * 8; i += 320) {
    int r = i >> 3, seg = i & 7;
    async_copy_b128(qBF + (grow + r) * kDM + h * kDH + seg * 8,
                    &qt[r * kLDT + seg * 8]);
  }
  if (tid < kMP) ksumS[tid] = ksum[bh * kMP + tid];
  if (tid < 64)  rowmax[tid] = (int)0xFF800000;
  wait_async0();
  __syncthreads();
  // xd
  f32x8 x[4][2];
  for (int mt = 0; mt < 4; ++mt) for (int nt = 0; nt < 2; ++nt) x[mt][nt] = zero8();
  #pragma unroll
  for (int k0 = 0; k0 < 64; k0 += 32) {
    #pragma unroll
    for (int mt = 0; mt < 4; ++mt) {
      bf16x16 a = load_frag(qt + (mt * 16) * kLDT + k0, kLDT, lane);
      #pragma unroll
      for (int nt = 0; nt < 2; ++nt) {
        bf16x16 bq = load_frag(projBF + (size_t)(w * 32 + nt * 16) * kDH + k0, kDH, lane);
        x[mt][nt] = wmma_bf16(a, bq, x[mt][nt]);
      }
    }
  }
  // per-row max over real features
  #pragma unroll
  for (int mt = 0; mt < 4; ++mt)
    #pragma unroll
    for (int nt = 0; nt < 2; ++nt) {
      int feat = w * 32 + nt * 16 + lc;
      if (feat < kMF) {
        #pragma unroll
        for (int r = 0; r < 8; ++r)
          atomicMaxFloatBits(&rowmax[mt * 16 + r + 8 * g], x[mt][nt][r]);
      }
    }
  __syncthreads();
  // qp epilogue -> qps (row-major)
  #pragma unroll
  for (int mt = 0; mt < 4; ++mt)
    #pragma unroll
    for (int nt = 0; nt < 2; ++nt) {
      int feat = w * 32 + nt * 16 + lc;
      #pragma unroll
      for (int r = 0; r < 8; ++r) {
        int srow = mt * 16 + r + 8 * g;
        float qp = 0.f;
        if (feat < kMF) {
          float mrow = __int_as_float(rowmax[srow]);
          qp = ratio * (__expf(x[mt][nt][r] - diagQ[bh * kS + s0 + srow] - mrow) + 1e-4f);
        }
        qps[srow * kLDQ + feat] = f2bf(qp);
      }
    }
  __syncthreads();
  if (tid < 64) {
    float dsum = 0.f;
    for (int m = 0; m < kMP; ++m) dsum += bf2f(qps[tid * kLDQ + m]) * ksumS[m];
    dinv[tid] = 1.f / dsum;
  }
  __syncthreads();
  // out = qps(64x320) @ ctxT(64x320)^T ; waves 0..7 (wave -> 16 rows x 32 cols)
  if (w < 8) {
    const int mt = w & 3, nh = w >> 2;
    f32x8 o[2]; o[0] = zero8(); o[1] = zero8();
    for (int k0 = 0; k0 < kMP; k0 += 32) {
      bf16x16 a = load_frag(qps + (size_t)(mt * 16) * kLDQ + k0, kLDQ, lane);
      #pragma unroll
      for (int nt = 0; nt < 2; ++nt) {
        bf16x16 bc = load_frag(ctxT + ((size_t)bh * kDH + nh * 32 + nt * 16) * kMP + k0, kMP, lane);
        o[nt] = wmma_bf16(a, bc, o[nt]);
      }
    }
    #pragma unroll
    for (int nt = 0; nt < 2; ++nt) {
      int dcol = nh * 32 + nt * 16 + lc;
      #pragma unroll
      for (int r = 0; r < 8; ++r) {
        int srow = mt * 16 + r + 8 * g;
        attnBF[(grow + srow) * kDM + h * kDH + dcol] = f2bf(o[nt][r] * dinv[srow]);
      }
    }
  }
}

// ---------------- residual + LayerNorm (row of 1024 per block) ----------------
__global__ __launch_bounds__(256) void ln_kernel(
    const float* __restrict__ X, const float* __restrict__ Y,
    const float* __restrict__ gw, const float* __restrict__ bw,
    float* __restrict__ outF, unsigned short* __restrict__ outB)
{
  __shared__ float red[256];
  const int row = blockIdx.x, t = threadIdx.x;
  const float* x = X + (size_t)row * kDM;
  const float* y = Y + (size_t)row * kDM;
  float v[4]; float s = 0.f;
  #pragma unroll
  for (int i = 0; i < 4; ++i) { v[i] = x[t + i * 256] + y[t + i * 256]; s += v[i]; }
  red[t] = s; __syncthreads();
  for (int st = 128; st > 0; st >>= 1) { if (t < st) red[t] += red[t + st]; __syncthreads(); }
  float mu = red[0] * (1.f / 1024.f);
  __syncthreads();
  float s2 = 0.f;
  #pragma unroll
  for (int i = 0; i < 4; ++i) { float d = v[i] - mu; s2 += d * d; }
  red[t] = s2; __syncthreads();
  for (int st = 128; st > 0; st >>= 1) { if (t < st) red[t] += red[t + st]; __syncthreads(); }
  float rs = rsqrtf(red[0] * (1.f / 1024.f) + 1e-5f);
  #pragma unroll
  for (int i = 0; i < 4; ++i) {
    int c = t + i * 256;
    float o = (v[i] - mu) * rs * gw[c] + bw[c];
    if (outF) outF[(size_t)row * kDM + c] = o;
    if (outB) outB[(size_t)row * kDM + c] = f2bf(o);
  }
}

// ---------------- host orchestration ----------------
extern "C" void kernel_launch(void* const* d_in, const int* in_sizes, int n_in,
                              void* d_out, int out_size, void* d_ws, size_t ws_size,
                              hipStream_t stream) {
  (void)in_sizes; (void)n_in; (void)out_size; (void)ws_size;
  const float* src = (const float*)d_in[0];
  const float* Wq  = (const float*)d_in[1];  const float* bq  = (const float*)d_in[2];
  const float* Wk  = (const float*)d_in[3];  const float* bk  = (const float*)d_in[4];
  const float* Wv  = (const float*)d_in[5];  const float* bv  = (const float*)d_in[6];
  const float* prj = (const float*)d_in[7];
  const float* Wo  = (const float*)d_in[8];  const float* bo  = (const float*)d_in[9];
  const float* W1  = (const float*)d_in[10]; const float* b1  = (const float*)d_in[11];
  const float* W2  = (const float*)d_in[12]; const float* b2  = (const float*)d_in[13];
  const float* g1  = (const float*)d_in[14]; const float* be1 = (const float*)d_in[15];
  const float* g2  = (const float*)d_in[16]; const float* be2 = (const float*)d_in[17];
  float* out = (float*)d_out;

  char* ws = (char*)d_ws; size_t off = 0;
  auto alloc = [&](size_t bytes) -> char* {
    char* p = ws + off; off += (bytes + 255) & ~(size_t)255; return p;
  };
  const size_t nTok = (size_t)kROWS * kDM;             // 16384*1024
  unsigned short* srcBF  = (unsigned short*)alloc(nTok * 2);
  unsigned short* WqBF   = (unsigned short*)alloc((size_t)kDM * kDM * 2);
  unsigned short* WkBF   = (unsigned short*)alloc((size_t)kDM * kDM * 2);
  unsigned short* WvBF   = (unsigned short*)alloc((size_t)kDM * kDM * 2);
  unsigned short* WoBF   = (unsigned short*)alloc((size_t)kDM * kDM * 2);
  unsigned short* W1BF   = (unsigned short*)alloc((size_t)kDFF * kDM * 2);
  unsigned short* W2BF   = (unsigned short*)alloc((size_t)kDM * kDFF * 2);
  unsigned short* projBF = (unsigned short*)alloc((size_t)kMP * kDH * 2);
  unsigned short* qBF    = (unsigned short*)alloc(nTok * 2);
  unsigned short* kBF    = (unsigned short*)alloc(nTok * 2);
  unsigned short* vBF    = (unsigned short*)alloc(nTok * 2);
  float* diagQ = (float*)alloc((size_t)64 * kS * 4);
  float* diagK = (float*)alloc((size_t)64 * kS * 4);
  float* kmaxB = (float*)alloc(64 * 4);
  float* ksumB = (float*)alloc((size_t)64 * kMP * 4);
  float* ctxB  = (float*)alloc((size_t)64 * kMP * kDH * 4);
  unsigned short* ctxT = (unsigned short*)alloc((size_t)64 * kDH * kMP * 2);
  unsigned short* attnBF = (unsigned short*)alloc(nTok * 2);
  float* attnF = (float*)alloc(nTok * 4);
  float* x1F   = (float*)alloc(nTok * 4);
  unsigned short* x1BF = (unsigned short*)alloc(nTok * 2);
  unsigned short* h1BF = (unsigned short*)alloc((size_t)kROWS * kDFF * 2);
  float* f2F   = (float*)alloc(nTok * 4);

  auto cvt = [&](const float* in, unsigned short* o, size_t n) {
    f32bf_kernel<<<(unsigned)((n + 255) / 256), 256, 0, stream>>>(in, o, (int)n);
  };
  cvt(src, srcBF, nTok);
  cvt(Wq, WqBF, (size_t)kDM * kDM);  cvt(Wk, WkBF, (size_t)kDM * kDM);
  cvt(Wv, WvBF, (size_t)kDM * kDM);  cvt(Wo, WoBF, (size_t)kDM * kDM);
  cvt(W1, W1BF, (size_t)kDFF * kDM); cvt(W2, W2BF, (size_t)kDM * kDFF);
  proj_kernel<<<(kMP * kDH + 255) / 256, 256, 0, stream>>>(prj, projBF);

  const int nctx = 64 * kMP * kDH;
  init_kernel<<<(nctx + 255) / 256, 256, 0, stream>>>(ctxB, ksumB, kmaxB, nctx, 64 * kMP);

  dim3 gQKV(kDM / 64, kROWS / 128);
  gemm_bf16<<<gQKV, 256, 0, stream>>>(srcBF, WqBF, bq, nullptr, qBF, kROWS, kDM, kDM, kDM, kDM, kDM, 4);
  gemm_bf16<<<gQKV, 256, 0, stream>>>(srcBF, WkBF, bk, nullptr, kBF, kROWS, kDM, kDM, kDM, kDM, kDM, 4);
  gemm_bf16<<<gQKV, 256, 0, stream>>>(srcBF, WvBF, bv, nullptr, vBF, kROWS, kDM, kDM, kDM, kDM, kDM, 4);

  diag_kernel<<<(64 * kS) / 256, 256, 0, stream>>>(qBF, diagQ);
  diag_kernel<<<(64 * kS) / 256, 256, 0, stream>>>(kBF, diagK);
  kmax_kernel<<<dim3(64, kS / 256), 256, 0, stream>>>(kBF, projBF, kmaxB);

  ctx_kernel<<<dim3(64, 8), 320, 0, stream>>>(kBF, vBF, projBF, diagK, kmaxB, ctxB, ksumB);
  ctxT_kernel<<<(nctx + 255) / 256, 256, 0, stream>>>(ctxB, ctxT);
  attn_kernel<<<dim3(64, kS / 64), 320, 0, stream>>>(qBF, projBF, ctxT, diagQ, ksumB, attnBF);

  gemm_bf16<<<gQKV, 256, 0, stream>>>(attnBF, WoBF, bo, attnF, nullptr, kROWS, kDM, kDM, kDM, kDM, kDM, 2);
  ln_kernel<<<kROWS, 256, 0, stream>>>(src, attnF, g1, be1, x1F, x1BF);

  dim3 gFF1(kDFF / 64, kROWS / 128);
  gemm_bf16<<<gFF1, 256, 0, stream>>>(x1BF, W1BF, b1, nullptr, h1BF, kROWS, kDFF, kDM, kDM, kDM, kDFF, 4 | 1);
  gemm_bf16<<<gQKV, 256, 0, stream>>>(h1BF, W2BF, b2, f2F, nullptr, kROWS, kDM, kDFF, kDFF, kDFF, kDM, 2);
  ln_kernel<<<kROWS, 256, 0, stream>>>(x1F, f2F, g2, be2, out, nullptr);
}
